// SNN_12567074308281
// MI455X (gfx1250) — matrix operand order repaired and checked by reading, not compiled, not run on hardware
//
#include <hip/hip_runtime.h>

// CDNA5 / gfx1250, wave32. One workgroup (128 threads = 4 waves) per batch row.
// Phase 1: stage W as f16 in LDS (vectorized, branch-free).
// Phase 2: WMMA GEMM drive[t,o] into LDS (branch-free clamped loads).
// Phase 3: 10-lane LIF scan; exact early exit at first spike (resets can
//          never affect the first-spike time).

typedef __attribute__((ext_vector_type(16))) _Float16 v16h;
typedef __attribute__((ext_vector_type(4)))  _Float16 v4h;
typedef __attribute__((ext_vector_type(8)))  _Float16 v8h;
typedef __attribute__((ext_vector_type(8)))  float    v8f;

#define STEPS   200
#define NDRV    199     // steps-1 drive values (t = 0..198)
#define IN_DIM  784
#define OUT_DIM 10
#define KPAD    800     // 784 padded to 25*32
#define KSTEPS  25
#define TTILES  13      // ceil(199/16)
#define TROWS   208     // 13*16
#define NWAVES  4
#define BLOCK   128

__device__ __forceinline__ v8f wmma_f16(const v16h a, const v16h b, v8f c) {
    return __builtin_amdgcn_wmma_f32_16x16x32_f16(
        /*neg_a=*/false, a, /*neg_b=*/false, b,
        /*c_mod=*/(short)0, c, /*reuse_a=*/false, /*reuse_b=*/false);
}

__global__ __launch_bounds__(BLOCK)
void snn_fst_kernel(const float* __restrict__ x,   // [512][784][200]
                    const float* __restrict__ w,   // [10][784]
                    float* __restrict__ out)       // [512][10]
{
    __shared__ _Float16 wh[16 * KPAD];   // f16 weights, [o(16)][i(800)], 25.6 KB
    __shared__ float    drv[TROWS * 16]; // drive[t][o], 13.3 KB

    const int tid = threadIdx.x;
    const int b   = blockIdx.x;

    // ---- Phase 1: stage weights into LDS as f16, zero-padded, branch-free ----
    #pragma unroll
    for (int it = 0; it < (16 * KPAD) / (BLOCK * 4); ++it) {   // 25 chunks
        const int idx = (it * BLOCK + tid) * 4;                // multiple of 4
        const int o   = idx / KPAD;
        const int i   = idx - o * KPAD;
        // chunks never straddle the i=784 boundary (784 % 4 == 0)
        const bool ok = (o < OUT_DIM) && (i <= IN_DIM - 4);
        const int  off = ok ? (o * IN_DIM + i) : 0;            // clamped, in-bounds
        float4 v = *(const float4*)&w[off];
        const float s = ok ? 1.0f : 0.0f;                      // mask via select+mul
        v4h hv;
        hv.x = (_Float16)(v.x * s);
        hv.y = (_Float16)(v.y * s);
        hv.z = (_Float16)(v.z * s);
        hv.w = (_Float16)(v.w * s);
        *(v4h*)&wh[idx] = hv;                                  // ds_store_b64
    }
    __syncthreads();

    // ---- Phase 2: WMMA GEMM: drive[t,o] = sum_i x[b,i,t] * w[o,i] ----
    const int lane = tid & 31;
    const int wave = tid >> 5;
    const int m    = lane & 15;   // A row (t within tile) == B/D column index
    const int h    = lane >> 4;   // half-wave
    const int kb   = h * 8;       // A K-offset base per ISA layout

    const float* __restrict__ xb = x + (size_t)b * IN_DIM * STEPS;

    for (int tile = wave; tile < TTILES; tile += NWAVES) {
        const int t0 = tile * 16;
        // Address clamp instead of load masking: x[b][i][199] is always
        // in-bounds; clamped lanes only write drv rows >= 199, never read.
        int t = t0 + m;
        if (t > NDRV) t = NDRV;
        const float* __restrict__ xrow = xb + t;   // index with i*STEPS

        v8f c = {};

        // Main K loop: i0 = 0..736 -> every i < 784, no guards at all.
        #pragma unroll 2
        for (int ki = 0; ki < KSTEPS - 1; ++ki) {
            const int i0 = ki * 32;
            float f[16];
            #pragma unroll
            for (int j = 0; j < 8; ++j) {
                f[j]     = __builtin_nontemporal_load(&xrow[(size_t)(i0 + kb + j)      * STEPS]);
                f[8 + j] = __builtin_nontemporal_load(&xrow[(size_t)(i0 + 16 + kb + j) * STEPS]);
            }
            v16h a;
            #pragma unroll
            for (int j = 0; j < 16; ++j) a[j] = (_Float16)f[j];

            const v8h blo = *(const v8h*)&wh[m * KPAD + i0 + 8 * h];
            const v8h bhi = *(const v8h*)&wh[m * KPAD + i0 + 16 + 8 * h];
            const v16h bf = __builtin_shufflevector(blo, bhi, 0, 1, 2, 3, 4, 5, 6, 7, 8, 9, 10, 11, 12, 13, 14, 15);
            c = wmma_f16(a, bf, c);
        }

        // Tail K-step (ki=24, i0=768): a[0..7] covers k<=15 (i<=783, in
        // bounds); a[8..15] covers k>=16 (i>=784) -> compile-time zero.
        {
            const int i0 = (KSTEPS - 1) * 32;   // 768
            float f[8];
            #pragma unroll
            for (int j = 0; j < 8; ++j)
                f[j] = __builtin_nontemporal_load(&xrow[(size_t)(i0 + kb + j) * STEPS]);
            v16h a;
            #pragma unroll
            for (int j = 0; j < 8; ++j) {
                a[j]     = (_Float16)f[j];
                a[8 + j] = (_Float16)0.0f;
            }
            const v8h blo = *(const v8h*)&wh[m * KPAD + i0 + 8 * h];
            const v8h bhi = *(const v8h*)&wh[m * KPAD + i0 + 16 + 8 * h];
            const v16h bf = __builtin_shufflevector(blo, bhi, 0, 1, 2, 3, 4, 5, 6, 7, 8, 9, 10, 11, 12, 13, 14, 15);
            c = wmma_f16(a, bf, c);
        }

        // D layout: VGPR r, lane L -> D[r + 8h][L&15]
        #pragma unroll
        for (int r = 0; r < 8; ++r)
            drv[(t0 + r + h * 8) * 16 + m] = c[r];
    }
    __syncthreads();

    // ---- Phase 3: LIF first-spike scan (10 lanes, exact early exit) ----
    // Before the first spike no reset has occurred, and post-spike dynamics
    // never affect the first-spike index, so breaking at the first crossing
    // is bit-exact w.r.t. the reference recurrence.
    if (tid < OUT_DIM) {
        const int o = tid;
        const float a_m = 1.0f - 0.1f / 20.0f;  // 0.995
        const float g_m = 0.1f / 20.0f;         // 0.005
        const float a_s = 1.0f - 0.1f / 5.0f;   // 0.98
        float V = 0.0f, I = 0.0f;
        float ans = (float)(STEPS - 1);         // 199 if silent
        for (int t = 0; t < NDRV; ++t) {
            const float Vn = a_m * V + g_m * I; // uses old V, old I
            I = a_s * I + drv[t * 16 + o];
            if (Vn > 1.0f) { ans = (float)(t + 1); break; }
            V = Vn;
        }
        out[b * OUT_DIM + o] = ans;
    }
}

extern "C" void kernel_launch(void* const* d_in, const int* in_sizes, int n_in,
                              void* d_out, int out_size, void* d_ws, size_t ws_size,
                              hipStream_t stream) {
    const float* x = (const float*)d_in[0];   // [512*784*200] f32
    const float* w = (const float*)d_in[1];   // [10*784] f32
    float* out     = (float*)d_out;           // [512*10] f32
    (void)in_sizes; (void)n_in; (void)out_size; (void)d_ws; (void)ws_size;

    snn_fst_kernel<<<dim3(512), dim3(BLOCK), 0, stream>>>(x, w, out);
}
